// SlotAttention_31731218382850
// MI455X (gfx1250) — matrix operand order repaired and checked by reading, not compiled, not run on hardware
//
#include <hip/hip_runtime.h>
#include <hip/hip_bf16.h>
#include <stdint.h>

typedef __attribute__((ext_vector_type(16))) _Float16 v16h;
typedef __attribute__((ext_vector_type(8)))  float    v8f;
typedef __attribute__((ext_vector_type(4)))  unsigned int u32x4;
typedef __attribute__((ext_vector_type(8)))  int      i32x8;
typedef __attribute__((ext_vector_type(4)))  int      i32x4;

#define BATCH 32
#define NSLOT 7
#define ZD    64
#define NPIX  4096        // 64*64
#define GDIM  70
#define BK    224         // BATCH*NSLOT
#define SCALE 0.125f      // 64^-0.5
#define EPSA  1e-8f

#if __has_builtin(__builtin_amdgcn_tensor_load_to_lds) && __has_builtin(__builtin_amdgcn_s_wait_tensorcnt)
#define USE_TDM 1
#else
#define USE_TDM 0
#endif

// ---------------------------------------------------------------------------
// WMMA fragment helpers (16x16x32 f16, wave32). A is 16(M)x32(K) row-major in
// LDS with stride `ld`; B tile stored as 16(N)x32(K) row-major.
// ---------------------------------------------------------------------------
__device__ __forceinline__ v16h frag_a(const _Float16* p, int ld) {
  int lane = threadIdx.x & 31;
  int m = lane & 15;
  int koff = (lane & 16) ? 8 : 0;
  v16h f;
#pragma unroll
  for (int i = 0; i < 8; ++i) {
    int k = ((i < 4) ? (2 * i) : (16 + 2 * (i - 4))) + koff;
    f[2 * i]     = p[m * ld + k];
    f[2 * i + 1] = p[m * ld + k + 1];
  }
  return f;
}

__device__ __forceinline__ v16h frag_b(const _Float16* p, int ld) {
  int lane = threadIdx.x & 31;
  int n = lane & 15;
  int koff = (lane & 16) ? 8 : 0;
  v16h f;
#pragma unroll
  for (int i = 0; i < 8; ++i) {
    int k = ((i < 4) ? (2 * i) : (16 + 2 * (i - 4))) + koff;
    f[2 * i]     = p[n * ld + k];
    f[2 * i + 1] = p[n * ld + k + 1];
  }
  return f;
}

__device__ __forceinline__ v8f wmma_fp16(v16h a, v16h b, v8f c) {
  return __builtin_amdgcn_wmma_f32_16x16x32_f16(false, a, false, b, (short)0, c,
                                                false, false);
}

#if USE_TDM
// Issue a TDM load of a dense 2D f16 tile (rows x cols, row stride = cols)
// from global memory into LDS, then the caller waits on TENSORcnt.
__device__ __forceinline__ void tdm_load_tile_f16(const _Float16* gsrc, void* ldst,
                                                  unsigned rows, unsigned cols) {
  unsigned long long ga = (unsigned long long)(uintptr_t)gsrc;
  unsigned int lds = (unsigned int)(uintptr_t)ldst;
  u32x4 g0;
  g0[0] = 1u;                                           // count=1
  g0[1] = lds;                                          // lds_addr
  g0[2] = (unsigned int)(ga & 0xFFFFFFFFu);             // global_addr lo
  g0[3] = (unsigned int)((ga >> 32) & 0x1FFFFFFu) | (2u << 30);  // hi | type=2
  i32x8 g1;
  g1[0] = 0x10000;                   // data_size=1 -> 2-byte elements
  g1[1] = (int)((cols & 0xFFFFu) << 16);   // tensor_dim0[15:0] @ bits[63:48]
  g1[2] = (int)((cols >> 16) | ((rows & 0xFFFFu) << 16)); // td0 hi | td1 lo
  g1[3] = (int)((rows >> 16) | ((cols & 0xFFFFu) << 16)); // td1 hi | tile_dim0
  g1[4] = (int)(rows & 0xFFFFu);     // tile_dim1 (tile_dim2 = 0)
  g1[5] = (int)cols;                 // tensor_dim0_stride lo32
  g1[6] = 0;
  g1[7] = 0;
  i32x4 gz4;
  gz4[0] = 0; gz4[1] = 0; gz4[2] = 0; gz4[3] = 0;
#if __clang_major__ >= 23
  i32x8 gz8;
  gz8[0] = 0; gz8[1] = 0; gz8[2] = 0; gz8[3] = 0;
  gz8[4] = 0; gz8[5] = 0; gz8[6] = 0; gz8[7] = 0;
  __builtin_amdgcn_tensor_load_to_lds(g0, g1, gz4, gz4, gz8, 0);
#else
  __builtin_amdgcn_tensor_load_to_lds(g0, g1, gz4, gz4, 0);
#endif
}
#endif

// ---------------------------------------------------------------------------
// Weight packing: conv OIHW f32 -> f16 [CoutPad][(ky*KS+kx)*Cin + c]
// ---------------------------------------------------------------------------
__global__ void pack_conv_w(const float* __restrict__ w, _Float16* __restrict__ dst,
                            int Cout, int CoutPad, int Cin, int KS) {
  int Ktot = Cin * KS * KS;
  int i = blockIdx.x * 256 + threadIdx.x;
  if (i >= CoutPad * Ktot) return;
  int o = i / Ktot, kv = i - o * Ktot;
  int c = kv % Cin;
  int rest = kv / Cin;
  int kx = rest % KS, ky = rest / KS;
  float v = 0.f;
  if (o < Cout) v = w[((o * Cin + c) * KS + ky) * KS + kx];
  dst[i] = (_Float16)v;
}

// [In][Out] f32 -> [Out][In] f16
__global__ void pack_mat_T(const float* __restrict__ w, _Float16* __restrict__ dst,
                           int In, int Out) {
  int i = blockIdx.x * 256 + threadIdx.x;
  if (i >= In * Out) return;
  int o = i / In, d = i - o * In;
  dst[i] = (_Float16)w[d * Out + o];
}

// inputs NCHW f32 -> NHWC f16
__global__ void nchw_to_nhwc_h(const float* __restrict__ x, _Float16* __restrict__ y,
                               int Nimg, int Cc, int HW) {
  long tot = (long)Nimg * Cc * HW;
  long i = (long)blockIdx.x * 256 + threadIdx.x;
  if (i >= tot) return;
  long n = i / ((long)Cc * HW);
  long rem = i - n * Cc * HW;
  int p = (int)(rem / Cc), c = (int)(rem - (long)p * Cc);
  y[i] = (_Float16)x[(n * Cc + c) * HW + p];
}

// decoder positional grid: pe[(i*G+j)*64+z]
__global__ void pe_dec_kernel(const float* __restrict__ dpw, const float* __restrict__ dpb,
                              float* __restrict__ pe, int Gd) {
  int i = blockIdx.x * 256 + threadIdx.x;
  int tot = Gd * Gd * 64;
  if (i >= tot) return;
  int z = i & 63;
  int ij = i >> 6;
  int jj = ij % Gd, ii = ij / Gd;
  float u = (float)ii / (float)(Gd - 1);
  float t = (float)jj / (float)(Gd - 1);
  pe[i] = (1.f - u) * dpw[z] + u * dpw[64 + z] + t * dpw[128 + z] +
          (1.f - t) * dpw[192 + z] + dpb[z];
}

// ---------------------------------------------------------------------------
// Generic implicit-GEMM conv (NHWC f16 in/out). mode==1: input is virtual
// broadcast slots[n][c] + pe[(y*Win+x)*64+c] (decoder dc1, no zb tensor).
// ---------------------------------------------------------------------------
__global__ void conv_wmma_kernel(const _Float16* __restrict__ wpack,
                                 const float* __restrict__ bias,
                                 const _Float16* __restrict__ in,
                                 const float* __restrict__ slots,
                                 const float* __restrict__ pe,
                                 _Float16* __restrict__ outH, float* __restrict__ outF,
                                 int Cin, int CoutReal, int KS, int pad,
                                 int Hin, int Win, int Hout, int Wout,
                                 int Ktot, int relu, int mode, int storeF32) {
  const int wave = threadIdx.x >> 5;
  const int lane = threadIdx.x & 31;
  const int n = blockIdx.z;
  const int ct = blockIdx.y;
  const int HW = Hout * Wout;
  const int pos0 = blockIdx.x * 64 + wave * 16;

  __shared__ _Float16 sA[16 * 32];
  __shared__ _Float16 sB[4][16 * 32];

  v8f acc = {0.f, 0.f, 0.f, 0.f, 0.f, 0.f, 0.f, 0.f};

  for (int k0 = 0; k0 < Ktot; k0 += 32) {
    __syncthreads();
    // prefetch next K chunk of packed weights (gfx1250 global_prefetch_b8)
    if (k0 + 32 < Ktot) {
      __builtin_prefetch(&wpack[(ct * 16 + (threadIdx.x & 15)) * Ktot + k0 + 32], 0, 0);
    }
    // stage A: 16 couts x 32 k
    for (int i = threadIdx.x; i < 512; i += 128) {
      int m = i >> 5, kk = i & 31;
      int kg = k0 + kk;
      sA[i] = (kg < Ktot) ? wpack[(ct * 16 + m) * Ktot + kg] : (_Float16)0.0f;
    }
    // stage B (im2col): 16 positions x 32 k, per wave
    for (int i = lane; i < 512; i += 32) {
      int r = i >> 5, kk = i & 31;
      int kg = k0 + kk;
      _Float16 val = (_Float16)0.0f;
      int posi = pos0 + r;
      if (posi < HW && kg < Ktot) {
        int oy = posi / Wout, ox = posi - oy * Wout;
        int cc = kg % Cin;
        int rest = kg / Cin;
        int kx = rest % KS, ky = rest / KS;
        int iy = oy + ky - pad, ix = ox + kx - pad;
        if (iy >= 0 && iy < Hin && ix >= 0 && ix < Win) {
          if (mode == 0) {
            val = in[((long)n * Hin * Win + (long)iy * Win + ix) * Cin + cc];
          } else {
            val = (_Float16)(slots[n * 64 + cc] + pe[((long)iy * Win + ix) * 64 + cc]);
          }
        }
      }
      sB[wave][i] = val;
    }
    __syncthreads();
    v16h a = frag_a(sA, 32);
    v16h b = frag_b(sB[wave], 32);
    acc = wmma_fp16(a, b, acc);
  }

  // D layout: lane -> N = lane&15; vgpr v -> M = v + 8*(lane>>4)
  int nn = lane & 15;
  int mbase = (lane & 16) ? 8 : 0;
  int posi = pos0 + nn;
  if (posi < HW) {
#pragma unroll
    for (int v2 = 0; v2 < 8; ++v2) {
      int cout = ct * 16 + mbase + v2;
      if (cout < CoutReal) {
        float r = acc[v2] + bias[cout];
        if (relu) r = r > 0.f ? r : 0.f;
        long idx = ((long)n * HW + posi) * CoutReal + cout;
        if (storeF32) outF[idx] = r;
        else outH[idx] = (_Float16)r;
      }
    }
  }
}

// ---------------------------------------------------------------------------
// WMMA GEMM: Y[N][64] = X[N][64] * W^T (Wp packed [out][in] f16), +bias, relu.
// Weight matrix (8KB, exact 64x64 f16 tile) is staged via the Tensor Data
// Mover (tensor_load_to_lds + s_wait_tensorcnt) when available.
// ---------------------------------------------------------------------------
__global__ void gemm64_wmma_kernel(const _Float16* __restrict__ X,
                                   const _Float16* __restrict__ Wp,
                                   const float* __restrict__ bias,
                                   _Float16* __restrict__ Y, long Nrows, int relu) {
  const int wave = threadIdx.x >> 5;
  const int lane = threadIdx.x & 31;
  __shared__ _Float16 sW[64 * 64];
  __shared__ _Float16 sX[4][16 * 64];

#if USE_TDM
  tdm_load_tile_f16(Wp, (void*)&sW[0], 64u, 64u);
#else
  for (int i = threadIdx.x; i < 4096; i += 128) sW[i] = Wp[i];
#endif

  // stage 16 rows of X per wave; each lane copies 32 contiguous halves (64B)
  long r0 = ((long)blockIdx.x * 4 + wave) * 16;
  {
    int rr = lane >> 1;
    int half = (lane & 1) * 32;
    long row = r0 + rr;
    _Float16* dst = &sX[wave][rr * 64 + half];
    if (row < Nrows) {
      const _Float16* src = &X[row * 64 + half];
#pragma unroll
      for (int j = 0; j < 32; ++j) dst[j] = src[j];
    } else {
#pragma unroll
      for (int j = 0; j < 32; ++j) dst[j] = (_Float16)0.0f;
    }
  }
#if USE_TDM
  __builtin_amdgcn_s_wait_tensorcnt(0);
#endif
  __syncthreads();

  int nn = lane & 15;
  int mbase = (lane & 16) ? 8 : 0;
  long row = r0 + nn;
#pragma unroll
  for (int mt = 0; mt < 4; ++mt) {
    v8f acc = {0.f, 0.f, 0.f, 0.f, 0.f, 0.f, 0.f, 0.f};
#pragma unroll
    for (int k0 = 0; k0 < 64; k0 += 32) {
      v16h a = frag_a(&sW[(mt * 16) * 64 + k0], 64);
      v16h b = frag_b(&sX[wave][k0], 64);
      acc = wmma_fp16(a, b, acc);
    }
    if (row < Nrows) {
#pragma unroll
      for (int v2 = 0; v2 < 8; ++v2) {
        int cout = mt * 16 + mbase + v2;
        float f = acc[v2] + (bias ? bias[cout] : 0.f);
        if (relu) f = f > 0.f ? f : 0.f;
        Y[row * 64 + cout] = (_Float16)f;
      }
    }
  }
}

// ---------------------------------------------------------------------------
// pos-embed add + LayerNorm over 64ch: one 64-thread block per position
// ---------------------------------------------------------------------------
__global__ void posln_kernel(const _Float16* __restrict__ x, const float* __restrict__ epw,
                             const float* __restrict__ epb, const float* __restrict__ g,
                             const float* __restrict__ bta, _Float16* __restrict__ y) {
  int pos = blockIdx.x;
  int c = threadIdx.x;
  int p = pos & (NPIX - 1);
  int ii = p >> 6, jj = p & 63;
  float u = (float)ii * (1.f / 63.f), t = (float)jj * (1.f / 63.f);
  float val = (float)x[(long)pos * 64 + c] + (1.f - u) * epw[c] + u * epw[64 + c] +
              t * epw[128 + c] + (1.f - t) * epw[192 + c] + epb[c];
  __shared__ float buf[64];
  buf[c] = val;
  __syncthreads();
  for (int o = 32; o >= 1; o >>= 1) { if (c < o) buf[c] += buf[c + o]; __syncthreads(); }
  float mean = buf[0] * (1.f / 64.f);
  __syncthreads();
  float dv = val - mean;
  buf[c] = dv * dv;
  __syncthreads();
  for (int o = 32; o >= 1; o >>= 1) { if (c < o) buf[c] += buf[c + o]; __syncthreads(); }
  float var = buf[0] * (1.f / 64.f);
  y[(long)pos * 64 + c] = (_Float16)(dv * rsqrtf(var + 1e-5f) * g[c] + bta[c]);
}

// slots init: loc + softplus(sp) * noise
__global__ void slot_init_kernel(const float* __restrict__ noise,
                                 const float* __restrict__ init_slots,
                                 float* __restrict__ slots) {
  int i = blockIdx.x * 256 + threadIdx.x;
  if (i >= BK * ZD) return;
  int z = i & 63;
  float sp = init_slots[64 + z];
  slots[i] = init_slots[z] + log1pf(expf(sp)) * noise[i];
}

// q = LN(slots) @ q_w * SCALE, one 64-thread block per (b,k)
__global__ void qproj_kernel(const float* __restrict__ slots, const float* __restrict__ g,
                             const float* __restrict__ b, const float* __restrict__ qw,
                             float* __restrict__ q) {
  int bk = blockIdx.x;
  int c = threadIdx.x;
  __shared__ float buf[64], ln[64];
  float v = slots[bk * 64 + c];
  buf[c] = v;
  __syncthreads();
  for (int o = 32; o >= 1; o >>= 1) { if (c < o) buf[c] += buf[c + o]; __syncthreads(); }
  float mean = buf[0] * (1.f / 64.f);
  __syncthreads();
  float dv = v - mean;
  buf[c] = dv * dv;
  __syncthreads();
  for (int o = 32; o >= 1; o >>= 1) { if (c < o) buf[c] += buf[c + o]; __syncthreads(); }
  float var = buf[0] * (1.f / 64.f);
  ln[c] = dv * rsqrtf(var + 1e-5f) * g[c] + b[c];
  __syncthreads();
  float acc = 0.f;
  for (int d = 0; d < 64; ++d) acc += ln[d] * qw[d * 64 + c];
  q[bk * 64 + c] = acc * SCALE;
}

// dots + softmax over slots (+EPS) + per-(b,k) key-sum via atomics
__global__ void attn_kernel(const float* __restrict__ q, const _Float16* __restrict__ key,
                            float* __restrict__ attn, float* __restrict__ rowsum) {
  int b = blockIdx.y;
  int n = blockIdx.x * 256 + threadIdx.x;
  __shared__ float sq[NSLOT * 64];
  for (int i = threadIdx.x; i < NSLOT * 64; i += 256) sq[i] = q[(long)b * NSLOT * 64 + i];
  __syncthreads();
  const _Float16* kr = &key[((long)b * NPIX + n) * 64];
  float kk[64];
#pragma unroll
  for (int z = 0; z < 64; ++z) kk[z] = (float)kr[z];
  float d[NSLOT];
  float mx = -1e30f;
#pragma unroll
  for (int k = 0; k < NSLOT; ++k) {
    float a = 0.f;
    for (int z = 0; z < 64; ++z) a += sq[k * 64 + z] * kk[z];
    d[k] = a;
    mx = a > mx ? a : mx;
  }
  float se = 0.f;
#pragma unroll
  for (int k = 0; k < NSLOT; ++k) { d[k] = expf(d[k] - mx); se += d[k]; }
  float inv = 1.f / se;
#pragma unroll
  for (int k = 0; k < NSLOT; ++k) {
    float a = d[k] * inv + EPSA;
    attn[((long)b * NSLOT + k) * NPIX + n] = a;
    atomicAdd(&rowsum[b * NSLOT + k], a);
  }
}

// updates[bk][z] = sum_n attn[bk][n] * v[b][n][z] / rowsum[bk]
__global__ void updates_kernel(const float* __restrict__ attn, const _Float16* __restrict__ vbuf,
                               const float* __restrict__ rowsum, float* __restrict__ upd) {
  int bk = blockIdx.x;
  int z = threadIdx.x;
  int b = bk / NSLOT;
  const float* a = &attn[(long)bk * NPIX];
  const _Float16* vb = &vbuf[(long)b * NPIX * 64];
  __shared__ float sa[256];
  float acc = 0.f;
  for (int n0 = 0; n0 < NPIX; n0 += 256) {
    __syncthreads();
    for (int i = threadIdx.x; i < 256; i += 64) sa[i] = a[n0 + i];
    __syncthreads();
    for (int j = 0; j < 256; ++j) acc += sa[j] * (float)vb[(long)(n0 + j) * 64 + z];
  }
  upd[bk * 64 + z] = acc / rowsum[bk];
}

// GRU cell (torch gate order r,z,n)
__global__ void gru_kernel(const float* __restrict__ upd, const float* __restrict__ slots,
                           const float* __restrict__ wih, const float* __restrict__ whh,
                           const float* __restrict__ bih, const float* __restrict__ bhh,
                           float* __restrict__ out) {
  int bk = blockIdx.x;
  int j = threadIdx.x;
  __shared__ float su[64], sh[64];
  su[j] = upd[bk * 64 + j];
  sh[j] = slots[bk * 64 + j];
  __syncthreads();
  float gi[3], gh[3];
#pragma unroll
  for (int p = 0; p < 3; ++p) {
    int row = p * 64 + j;
    float a = bih[row], c = bhh[row];
    for (int d = 0; d < 64; ++d) {
      a += su[d] * wih[row * 64 + d];
      c += sh[d] * whh[row * 64 + d];
    }
    gi[p] = a; gh[p] = c;
  }
  float r  = 1.f / (1.f + expf(-(gi[0] + gh[0])));
  float zg = 1.f / (1.f + expf(-(gi[1] + gh[1])));
  float nn = tanhf(gi[2] + r * gh[2]);
  out[bk * 64 + j] = (1.f - zg) * nn + zg * sh[j];
}

// slots += mlp2(relu(mlp1(LN(slots)))) -- 128 threads per (b,k)
__global__ void slotmlp_kernel(const float* __restrict__ sin, const float* __restrict__ g,
                               const float* __restrict__ b, const float* __restrict__ w1,
                               const float* __restrict__ b1, const float* __restrict__ w2,
                               const float* __restrict__ b2, float* __restrict__ sout) {
  int bk = blockIdx.x;
  int t = threadIdx.x;  // 0..127
  __shared__ float s[64], ln[64], hid[128], buf[64];
  if (t < 64) { s[t] = sin[bk * 64 + t]; buf[t] = sin[bk * 64 + t]; }
  __syncthreads();
  for (int o = 32; o >= 1; o >>= 1) { if (t < o) buf[t] += buf[t + o]; __syncthreads(); }
  float mean = buf[0] * (1.f / 64.f);
  __syncthreads();
  if (t < 64) { float dv = s[t] - mean; buf[t] = dv * dv; }
  __syncthreads();
  for (int o = 32; o >= 1; o >>= 1) { if (t < o) buf[t] += buf[t + o]; __syncthreads(); }
  float var = buf[0] * (1.f / 64.f);
  __syncthreads();
  if (t < 64) ln[t] = (s[t] - mean) * rsqrtf(var + 1e-5f) * g[t] + b[t];
  __syncthreads();
  float h = b1[t];
  for (int d = 0; d < 64; ++d) h += ln[d] * w1[d * 128 + t];
  hid[t] = h > 0.f ? h : 0.f;
  __syncthreads();
  if (t < 64) {
    float o = b2[t];
    for (int u = 0; u < 128; ++u) o += hid[u] * w2[u * 64 + t];
    sout[bk * 64 + t] = s[t] + o;
  }
}

// sigmoid / log-softmax over slots / recon / mse
__global__ void final_kernel(const float* __restrict__ inputs, const float* __restrict__ o4,
                             float* __restrict__ out) {
  const long XLOC = 1;
  const long LOGP = 1 + (long)BK * 3 * NPIX;
  long i = (long)blockIdx.x * 256 + threadIdx.x;
  if (i >= (long)BATCH * NPIX) return;
  int b = (int)(i >> 12);
  int yx = (int)(i & (NPIX - 1));
  float m[NSLOT];
  float mx = -1e30f;
#pragma unroll
  for (int k = 0; k < NSLOT; ++k) {
    m[k] = o4[(((long)(b * NSLOT + k) * NPIX) + yx) * 4 + 3];
    mx = m[k] > mx ? m[k] : mx;
  }
  float se = 0.f;
#pragma unroll
  for (int k = 0; k < NSLOT; ++k) se += expf(m[k] - mx);
  float lse = mx + logf(se);
  float recon[3] = {0.f, 0.f, 0.f};
#pragma unroll
  for (int k = 0; k < NSLOT; ++k) {
    float lp = m[k] - lse;
    out[LOGP + (long)(b * NSLOT + k) * NPIX + yx] = lp;
    float w = expf(lp);
#pragma unroll
    for (int c = 0; c < 3; ++c) {
      float s = 1.f / (1.f + expf(-o4[(((long)(b * NSLOT + k) * NPIX) + yx) * 4 + c]));
      out[XLOC + ((long)(b * NSLOT + k) * 3 + c) * NPIX + yx] = s;
      recon[c] += s * w;
    }
  }
  float sq = 0.f;
#pragma unroll
  for (int c = 0; c < 3; ++c) {
    float xo = (inputs[((long)b * 3 + c) * NPIX + yx] + 1.f) * 0.5f;
    float d = xo - recon[c];
    sq += d * d;
  }
  atomicAdd(out, sq * (1.f / (float)BATCH));
}

__global__ void copy_slots_kernel(const float* __restrict__ s, float* __restrict__ out) {
  const long SLOTS = 1 + (long)BK * 3 * NPIX + (long)BK * NPIX;
  int i = blockIdx.x * 256 + threadIdx.x;
  if (i < BK * ZD) out[SLOTS + i] = s[i];
}

// ---------------------------------------------------------------------------
extern "C" void kernel_launch(void* const* d_in, const int* in_sizes, int n_in,
                              void* d_out, int out_size, void* d_ws, size_t ws_size,
                              hipStream_t stream) {
  (void)in_sizes; (void)n_in; (void)out_size; (void)ws_size;
  const float* inputs    = (const float*)d_in[0];
  const float* noise     = (const float*)d_in[1];
  const float* enc_pos_w = (const float*)d_in[2];
  const float* enc_pos_b = (const float*)d_in[3];
  const float* ec1_w = (const float*)d_in[4];  const float* ec1_b = (const float*)d_in[5];
  const float* ec2_w = (const float*)d_in[6];  const float* ec2_b = (const float*)d_in[7];
  const float* ec3_w = (const float*)d_in[8];  const float* ec3_b = (const float*)d_in[9];
  const float* ec4_w = (const float*)d_in[10]; const float* ec4_b = (const float*)d_in[11];
  const float* ln1_g = (const float*)d_in[12]; const float* ln1_b = (const float*)d_in[13];
  const float* fc1_w = (const float*)d_in[14]; const float* fc1_b = (const float*)d_in[15];
  const float* fc2_w = (const float*)d_in[16]; const float* fc2_b = (const float*)d_in[17];
  const float* lns_g = (const float*)d_in[18]; const float* lns_b = (const float*)d_in[19];
  const float* lnf_g = (const float*)d_in[20]; const float* lnf_b = (const float*)d_in[21];
  const float* q_w   = (const float*)d_in[22];
  const float* k_w   = (const float*)d_in[23];
  const float* v_w   = (const float*)d_in[24];
  const float* mlp1_w = (const float*)d_in[25]; const float* mlp1_b = (const float*)d_in[26];
  const float* mlp2_w = (const float*)d_in[27]; const float* mlp2_b = (const float*)d_in[28];
  const float* gru_wih = (const float*)d_in[29]; const float* gru_whh = (const float*)d_in[30];
  const float* gru_bih = (const float*)d_in[31]; const float* gru_bhh = (const float*)d_in[32];
  const float* init_slots = (const float*)d_in[33];
  const float* dec_pos_w  = (const float*)d_in[34];
  const float* dec_pos_b  = (const float*)d_in[35];
  const float* dc1_w = (const float*)d_in[36]; const float* dc1_b = (const float*)d_in[37];
  const float* dc2_w = (const float*)d_in[38]; const float* dc2_b = (const float*)d_in[39];
  const float* dc3_w = (const float*)d_in[40]; const float* dc3_b = (const float*)d_in[41];
  const float* dc4_w = (const float*)d_in[42]; const float* dc4_b = (const float*)d_in[43];

  char* ws = (char*)d_ws;
  size_t off = 0;
  auto alloc = [&](size_t b) { size_t r = off; off += (b + 255) & ~(size_t)255; return r; };
  auto alg = [](size_t b) { return (b + 255) & ~(size_t)255; };

  size_t oWEC1 = alloc((size_t)64 * 75 * 2);
  size_t oWEC2 = alloc((size_t)64 * 1600 * 2);
  size_t oWEC3 = alloc((size_t)64 * 1600 * 2);
  size_t oWEC4 = alloc((size_t)64 * 1600 * 2);
  size_t oWFC1 = alloc((size_t)64 * 64 * 2);
  size_t oWFC2 = alloc((size_t)64 * 64 * 2);
  size_t oWKW  = alloc((size_t)64 * 64 * 2);
  size_t oWVW  = alloc((size_t)64 * 64 * 2);
  size_t oWDC1 = alloc((size_t)32 * 1600 * 2);
  size_t oWDC2 = alloc((size_t)32 * 800 * 2);
  size_t oWDC3 = alloc((size_t)32 * 800 * 2);
  size_t oWDC4 = alloc((size_t)16 * 288 * 2);
  size_t oXIN  = alloc((size_t)BATCH * NPIX * 3 * 2);
  size_t oSLOT = alloc((size_t)BK * 64 * 4);
  size_t oSLT2 = alloc((size_t)BK * 64 * 4);
  size_t oQ    = alloc((size_t)BK * 64 * 4);
  size_t oUPD  = alloc((size_t)BK * 64 * 4);
  size_t oRSUM = alloc((size_t)BK * 4);
  size_t oPED  = alloc((size_t)GDIM * GDIM * 64 * 4);

  size_t szXA = (size_t)BATCH * NPIX * 64 * 2;                 // f16 act
  size_t szATTN = (size_t)BATCH * NSLOT * NPIX * 4;
  size_t encBytes = 4 * alg(szXA) + alg(szATTN);
  size_t szO1 = (size_t)BK * 68 * 68 * 32 * 2;
  size_t szO2 = (size_t)BK * 66 * 66 * 32 * 2;
  size_t decBytes = alg(szO1) + alg(szO2);
  size_t oBIG = alloc(encBytes > decBytes ? encBytes : decBytes);

  _Float16* wpEC1 = (_Float16*)(ws + oWEC1);
  _Float16* wpEC2 = (_Float16*)(ws + oWEC2);
  _Float16* wpEC3 = (_Float16*)(ws + oWEC3);
  _Float16* wpEC4 = (_Float16*)(ws + oWEC4);
  _Float16* wpFC1 = (_Float16*)(ws + oWFC1);
  _Float16* wpFC2 = (_Float16*)(ws + oWFC2);
  _Float16* wpKW  = (_Float16*)(ws + oWKW);
  _Float16* wpVW  = (_Float16*)(ws + oWVW);
  _Float16* wpDC1 = (_Float16*)(ws + oWDC1);
  _Float16* wpDC2 = (_Float16*)(ws + oWDC2);
  _Float16* wpDC3 = (_Float16*)(ws + oWDC3);
  _Float16* wpDC4 = (_Float16*)(ws + oWDC4);
  _Float16* xin   = (_Float16*)(ws + oXIN);
  float* slots = (float*)(ws + oSLOT);
  float* slot2 = (float*)(ws + oSLT2);
  float* qbuf  = (float*)(ws + oQ);
  float* upd   = (float*)(ws + oUPD);
  float* rsum  = (float*)(ws + oRSUM);
  float* peDec = (float*)(ws + oPED);

  char* big = ws + oBIG;
  _Float16* xa   = (_Float16*)(big);
  _Float16* xb   = (_Float16*)(big + alg(szXA));
  _Float16* kbuf = (_Float16*)(big + 2 * alg(szXA));
  _Float16* vbuf = (_Float16*)(big + 3 * alg(szXA));
  float*    attn = (float*)(big + 4 * alg(szXA));
  _Float16* o1 = (_Float16*)(big);
  _Float16* o2 = (_Float16*)(big + alg(szO1));
  _Float16* o3 = o1;
  float*    o4 = (float*)(big + alg(szO1));

  float* out = (float*)d_out;
  auto cdiv = [](long a, long b) { return (int)((a + b - 1) / b); };

  // ---- weight packing / input conversion / decoder pos grid ----
  pack_conv_w<<<cdiv(64 * 75, 256), 256, 0, stream>>>(ec1_w, wpEC1, 64, 64, 3, 5);
  pack_conv_w<<<cdiv(64 * 1600, 256), 256, 0, stream>>>(ec2_w, wpEC2, 64, 64, 64, 5);
  pack_conv_w<<<cdiv(64 * 1600, 256), 256, 0, stream>>>(ec3_w, wpEC3, 64, 64, 64, 5);
  pack_conv_w<<<cdiv(64 * 1600, 256), 256, 0, stream>>>(ec4_w, wpEC4, 64, 64, 64, 5);
  pack_mat_T<<<cdiv(4096, 256), 256, 0, stream>>>(fc1_w, wpFC1, 64, 64);
  pack_mat_T<<<cdiv(4096, 256), 256, 0, stream>>>(fc2_w, wpFC2, 64, 64);
  pack_mat_T<<<cdiv(4096, 256), 256, 0, stream>>>(k_w, wpKW, 64, 64);
  pack_mat_T<<<cdiv(4096, 256), 256, 0, stream>>>(v_w, wpVW, 64, 64);
  pack_conv_w<<<cdiv(32 * 1600, 256), 256, 0, stream>>>(dc1_w, wpDC1, 32, 32, 64, 5);
  pack_conv_w<<<cdiv(32 * 800, 256), 256, 0, stream>>>(dc2_w, wpDC2, 32, 32, 32, 5);
  pack_conv_w<<<cdiv(32 * 800, 256), 256, 0, stream>>>(dc3_w, wpDC3, 32, 32, 32, 5);
  pack_conv_w<<<cdiv(16 * 288, 256), 256, 0, stream>>>(dc4_w, wpDC4, 4, 16, 32, 3);
  nchw_to_nhwc_h<<<cdiv((long)BATCH * 3 * NPIX, 256), 256, 0, stream>>>(inputs, xin, BATCH, 3, NPIX);
  pe_dec_kernel<<<cdiv(GDIM * GDIM * 64, 256), 256, 0, stream>>>(dec_pos_w, dec_pos_b, peDec, GDIM);

  // ---- encoder convs (WMMA implicit GEMM) ----
  dim3 cb(128);
  conv_wmma_kernel<<<dim3(64, 4, BATCH), cb, 0, stream>>>(wpEC1, ec1_b, xin, nullptr, nullptr,
      xa, nullptr, 3, 64, 5, 2, 64, 64, 64, 64, 75, 1, 0, 0);
  conv_wmma_kernel<<<dim3(64, 4, BATCH), cb, 0, stream>>>(wpEC2, ec2_b, xa, nullptr, nullptr,
      xb, nullptr, 64, 64, 5, 2, 64, 64, 64, 64, 1600, 1, 0, 0);
  conv_wmma_kernel<<<dim3(64, 4, BATCH), cb, 0, stream>>>(wpEC3, ec3_b, xb, nullptr, nullptr,
      xa, nullptr, 64, 64, 5, 2, 64, 64, 64, 64, 1600, 1, 0, 0);
  conv_wmma_kernel<<<dim3(64, 4, BATCH), cb, 0, stream>>>(wpEC4, ec4_b, xa, nullptr, nullptr,
      xb, nullptr, 64, 64, 5, 2, 64, 64, 64, 64, 1600, 1, 0, 0);

  // ---- pos-embed + LN, feature MLP, k/v projections (WMMA GEMMs) ----
  posln_kernel<<<BATCH * NPIX, 64, 0, stream>>>(xb, enc_pos_w, enc_pos_b, ln1_g, ln1_b, xa);
  long NR = (long)BATCH * NPIX;
  gemm64_wmma_kernel<<<cdiv(NR, 64), 128, 0, stream>>>(xa, wpFC1, fc1_b, xb, NR, 1);
  gemm64_wmma_kernel<<<cdiv(NR, 64), 128, 0, stream>>>(xb, wpFC2, fc2_b, xa, NR, 0);
  gemm64_wmma_kernel<<<cdiv(NR, 64), 128, 0, stream>>>(xa, wpKW, nullptr, kbuf, NR, 0);
  gemm64_wmma_kernel<<<cdiv(NR, 64), 128, 0, stream>>>(xa, wpVW, nullptr, vbuf, NR, 0);

  // ---- slot attention iterations ----
  slot_init_kernel<<<cdiv(BK * 64, 256), 256, 0, stream>>>(noise, init_slots, slots);
  for (int it = 0; it < 3; ++it) {
    qproj_kernel<<<BK, 64, 0, stream>>>(slots, lns_g, lns_b, q_w, qbuf);
    (void)hipMemsetAsync(rsum, 0, BK * sizeof(float), stream);
    attn_kernel<<<dim3(NPIX / 256, BATCH), 256, 0, stream>>>(qbuf, kbuf, attn, rsum);
    updates_kernel<<<BK, 64, 0, stream>>>(attn, vbuf, rsum, upd);
    gru_kernel<<<BK, 64, 0, stream>>>(upd, slots, gru_wih, gru_whh, gru_bih, gru_bhh, slot2);
    slotmlp_kernel<<<BK, 128, 0, stream>>>(slot2, lnf_g, lnf_b, mlp1_w, mlp1_b, mlp2_w, mlp2_b, slots);
  }

  // ---- decoder (dc1 fuses spatial-broadcast + pos-embed in im2col fetch) ----
  conv_wmma_kernel<<<dim3(cdiv(68 * 68, 64), 2, BK), cb, 0, stream>>>(wpDC1, dc1_b,
      nullptr, slots, peDec, o1, nullptr, 64, 32, 5, 1, 70, 70, 68, 68, 1600, 1, 1, 0);
  conv_wmma_kernel<<<dim3(cdiv(66 * 66, 64), 2, BK), cb, 0, stream>>>(wpDC2, dc2_b,
      o1, nullptr, nullptr, o2, nullptr, 32, 32, 5, 1, 68, 68, 66, 66, 800, 1, 0, 0);
  conv_wmma_kernel<<<dim3(64, 2, BK), cb, 0, stream>>>(wpDC3, dc3_b,
      o2, nullptr, nullptr, o3, nullptr, 32, 32, 5, 1, 66, 66, 64, 64, 800, 1, 0, 0);
  conv_wmma_kernel<<<dim3(64, 1, BK), cb, 0, stream>>>(wpDC4, dc4_b,
      o3, nullptr, nullptr, nullptr, o4, 32, 4, 3, 1, 64, 64, 64, 64, 288, 0, 0, 1);

  // ---- outputs ----
  (void)hipMemsetAsync(out, 0, sizeof(float), stream);  // mse accumulator
  final_kernel<<<cdiv((long)BATCH * NPIX, 256), 256, 0, stream>>>(inputs, o4, out);
  copy_slots_kernel<<<cdiv(BK * 64, 256), 256, 0, stream>>>(slots, out);
}
